// MultiheadAttention_70755291234430
// MI455X (gfx1250) — matrix operand order repaired and checked by reading, not compiled, not run on hardware
//
#include <hip/hip_runtime.h>
#include <hip/hip_bf16.h>

#define DEVFN __device__ __forceinline__

typedef __attribute__((ext_vector_type(16))) __bf16 v16bf;
typedef __attribute__((ext_vector_type(8)))  float  v8f;
typedef unsigned int v4u __attribute__((ext_vector_type(4)));
typedef int          v8i __attribute__((ext_vector_type(8)));
typedef int          v4i __attribute__((ext_vector_type(4)));

constexpr int Bc = 2, Sc = 2048, IND = 512, OUTD = 512;
constexpr int Hc = 8, KD = 64, VD = 64;

#if defined(__has_builtin)
#if __has_builtin(__builtin_amdgcn_tensor_load_to_lds) && \
    __has_builtin(__builtin_amdgcn_s_wait_tensorcnt)
#define HAVE_TDM 1
#endif
#endif

DEVFN unsigned short f32_to_bf16(float f) {
  unsigned int u = __float_as_uint(f);
  u += 0x7fffu + ((u >> 16) & 1u);   // round-to-nearest-even
  return (unsigned short)(u >> 16);
}

// A fragment: 16x32 bf16, M = lane%16, slots: half==0 -> K[c0..c0+15]
//                                            half==1 -> K[c0+16..c0+31]
DEVFN v16bf load_fragA_bf16(const unsigned short* base, int row0, int ld, int col0) {
  int lane = threadIdx.x & 31, half = lane >> 4, r = lane & 15;
  const unsigned short* p = base + (size_t)(row0 + r) * ld + col0 + half * 16;
  union { v16bf v; uint4 q[2]; } o;
  o.q[0] = *(const uint4*)p;
  o.q[1] = *(const uint4*)(p + 8);
  return o.v;
}

// B fragment: 32x16 (KxN) sourced from B^T rows (N-major storage baseT[n][k]).
// lane<16: N=lane, K=c0..c0+15 ; lane>=16: N=lane-16, K=c0+16..c0+31
DEVFN v16bf load_fragB_bf16(const unsigned short* baseT, int row0, int ld, int col0) {
  int lane = threadIdx.x & 31, half = lane >> 4, r = lane & 15;
  const unsigned short* p = baseT + (size_t)(row0 + r) * ld + col0 + half * 16;
  union { v16bf v; uint4 q[2]; } o;
  o.q[0] = *(const uint4*)p;
  o.q[1] = *(const uint4*)(p + 8);
  return o.v;
}

DEVFN v8f wmma_bf16(v16bf a, v16bf b, v8f c) {
  return __builtin_amdgcn_wmma_f32_16x16x32_bf16(false, a, false, b, (short)0, c,
                                                 false, false);
}

// ---------------------------------------------------------------- kernel 0a
// Transpose + bf16-convert weights: WkT/WvT [H][64][512], WoT [512][512]
__global__ __launch_bounds__(256) void prep_weights_kernel(
    const float* __restrict__ Wk, const float* __restrict__ Wv,
    const float* __restrict__ Wo, unsigned short* __restrict__ WkT,
    unsigned short* __restrict__ WvT, unsigned short* __restrict__ WoT) {
  int idx = blockIdx.x * blockDim.x + threadIdx.x;  // 0 .. 262143
  {
    int h = idx / (KD * IND);
    int rem = idx - h * (KD * IND);
    int o = rem / IND;
    int d = rem - o * IND;
    WkT[idx] = f32_to_bf16(Wk[((size_t)h * IND + d) * KD + o]);
    WvT[idx] = f32_to_bf16(Wv[((size_t)h * IND + d) * KD + o]);
  }
  {
    int o = idx / IND;
    int i = idx - o * IND;
    WoT[idx] = f32_to_bf16(Wo[(size_t)i * OUTD + o]);
  }
}

// ---------------------------------------------------------------- kernel 0b
// One-time x -> bf16 (so GEMM inner loops are pure load_b128 + wmma)
__global__ __launch_bounds__(256) void convert_x_kernel(
    const float* __restrict__ x, unsigned short* __restrict__ xb) {
  size_t i = (size_t)blockIdx.x * blockDim.x + threadIdx.x;  // over float4s
  float4 v = ((const float4*)x)[i];
  union { unsigned short s[4]; uint2 u; } o;
  o.s[0] = f32_to_bf16(v.x);
  o.s[1] = f32_to_bf16(v.y);
  o.s[2] = f32_to_bf16(v.z);
  o.s[3] = f32_to_bf16(v.w);
  ((uint2*)xb)[i] = o.u;
}

// ---------------------------------------------------------------- kernel 1
// K = x@Wk+bk (bf16 out), V = x@Wv+bv (f32 out). grid (B*H, S/16), 8 waves.
__global__ __launch_bounds__(256) void proj_kv_kernel(
    const unsigned short* __restrict__ xb, const unsigned short* __restrict__ WkT,
    const unsigned short* __restrict__ WvT, const float* __restrict__ bk,
    const float* __restrict__ bv, unsigned short* __restrict__ Kb,
    float* __restrict__ Vf) {
  int bh = blockIdx.x, b = bh / Hc, h = bh - b * Hc;
  int row0 = blockIdx.y * 16;
  int wave = threadIdx.x >> 5, lane = threadIdx.x & 31;
  int half = lane >> 4, r = lane & 15;
  bool isK = wave < 4;
  int n = isK ? wave : wave - 4;  // 16-col tile of the 64-wide head dim
  const unsigned short* WT = (isK ? WkT : WvT) + (size_t)h * KD * IND;
  const unsigned short* xrow = xb + (size_t)b * Sc * IND;

  v8f c = {};
  for (int kk = 0; kk < IND; kk += 32) {
    v16bf a = load_fragA_bf16(xrow, row0, IND, kk);
    v16bf bb = load_fragB_bf16(WT, n * 16, IND, kk);
    c = wmma_bf16(a, bb, c);
  }
  float bias = (isK ? bk : bv)[h * KD + n * 16 + r];
  size_t obase = (size_t)(b * Hc + h) * Sc + row0;
  if (isK) {
#pragma unroll
    for (int i = 0; i < 8; ++i) {
      int m = half * 8 + i;
      Kb[(obase + m) * KD + n * 16 + r] = f32_to_bf16(c[i] + bias);
    }
  } else {
#pragma unroll
    for (int i = 0; i < 8; ++i) {
      int m = half * 8 + i;
      Vf[(obase + m) * VD + n * 16 + r] = c[i] + bias;
    }
  }
}

// ---------------------------------------------------------------- kernel 2
// Per (b,h,16-query tile): TDM-stage query K tile to LDS, WMMA causal scores
// -> LDS strip, exact top-32 per row via wave32 iterative argmax, softmax
// over the 32, sparse gather of V.
__global__ __launch_bounds__(128) void attn_kernel(
    const unsigned short* __restrict__ Kb, const float* __restrict__ Vf,
    unsigned short* __restrict__ Yb) {
  extern __shared__ char smem_raw[];
  float* sc = (float*)smem_raw;                                    // 16 x 2048 f32
  unsigned short* kq = (unsigned short*)(smem_raw + 16 * Sc * 4);  // 16 x 64 bf16
  constexpr int SROW = Sc;
  const float NEG_INF = -__builtin_inff();

  int nqt = Sc / 16;
  int bh = blockIdx.x / nqt, qt = blockIdx.x - bh * nqt;
  int q0 = qt * 16;
  int b = bh / Hc, h = bh - b * Hc;
  const unsigned short* Kbh = Kb + (size_t)bh * Sc * KD;
  const float* Vbh = Vf + (size_t)bh * Sc * VD;
  int wave = threadIdx.x >> 5, lane = threadIdx.x & 31;
  int half = lane >> 4, r = lane & 15;
  const float scale = 0.125f;  // 1/sqrt(64)

#if defined(HAVE_TDM)
  // Tensor Data Mover: DMA the 16x64 bf16 query tile (K rows q0..q0+15) to LDS.
  if (wave == 0) {
    unsigned long long ga =
        (unsigned long long)(const void*)(Kbh + (size_t)q0 * KD);
    unsigned lds = (unsigned)(unsigned long long)(const void*)kq;  // addr[31:0]
    // D# group0: count=1 | lds_addr | global_addr[56:0] | type=2
    v4u g0 = {1u, lds, (unsigned)(ga & 0xffffffffu),
              ((unsigned)((ga >> 32) & 0x01ffffffu)) | (2u << 30)};
    // D# group1: data_size=2B; tensor 64 x 2048; tile 64 x 16; stride0=64
    v8i g1 = {(int)0x00010000,   // wg_mask=0, data_size=1 (2 bytes)
              (int)(64u << 16),  // tensor_dim0 = 64 (bits 79:48, low half)
              (int)(2048u << 16),// tensor_dim0 hi=0 | tensor_dim1 = 2048 (lo16)
              (int)(64u << 16),  // tensor_dim1 hi=0 | tile_dim0 = 64
              16,                // tile_dim1 = 16, tile_dim2 = 0
              64,                // tensor_dim0_stride = 64
              0, 0};
    v4i gz4 = {0, 0, 0, 0};
    v8i gz8 = {0, 0, 0, 0, 0, 0, 0, 0};
    __builtin_amdgcn_tensor_load_to_lds(g0, g1, gz4, gz4, gz8, 0);
    __builtin_amdgcn_s_wait_tensorcnt(0);
  }
  __syncthreads();
  v16bf a0 = load_fragA_bf16(kq, 0, KD, 0);
  v16bf a1 = load_fragA_bf16(kq, 0, KD, 32);
#else
  v16bf a0 = load_fragA_bf16(Kbh, q0, KD, 0);
  v16bf a1 = load_fragA_bf16(Kbh, q0, KD, 32);
#endif

  for (int jt = wave; jt <= qt; jt += 4) {
    int j0 = jt * 16;
    if (jt + 4 <= qt)  // prefetch next key tile (global_prefetch_b8)
      __builtin_prefetch(Kbh + (size_t)(j0 + 64) * KD + lane * 4, 0, 0);
    v16bf b0 = load_fragB_bf16(Kbh, j0, KD, 0);
    v16bf b1 = load_fragB_bf16(Kbh, j0, KD, 32);
    v8f c = {};
    c = wmma_bf16(a0, b0, c);
    c = wmma_bf16(a1, b1, c);
#pragma unroll
    for (int i = 0; i < 8; ++i) {
      int m = half * 8 + i;
      bool keep = (j0 + r) <= (q0 + m);  // causal
      sc[m * SROW + j0 + r] = keep ? c[i] * scale : NEG_INF;
    }
  }
  __syncthreads();

  int L = q0 + 16;  // valid score columns for this tile
  for (int rr = wave; rr < 16; rr += 4) {
    const float* row = sc + rr * SROW;
    // 32 iterations of wave-wide argmax under total order (value desc, idx asc)
    float pv = __builtin_inff();
    int pi = -1;
    float myv = NEG_INF;
    int myi = 0;
    for (int t = 0; t < 32; ++t) {
      float bestv = NEG_INF;
      int besti = 0x7fffffff;
      for (int j = lane; j < L; j += 32) {
        float vv = row[j];
        bool after = (vv < pv) || (vv == pv && j > pi);
        if (after && (vv > bestv || (vv == bestv && j < besti))) {
          bestv = vv;
          besti = j;
        }
      }
      for (int off = 16; off; off >>= 1) {
        float ov = __shfl_xor(bestv, off, 32);
        int oi = __shfl_xor(besti, off, 32);
        if (ov > bestv || (ov == bestv && oi < besti)) {
          bestv = ov;
          besti = oi;
        }
      }
      if (lane == t) { myv = bestv; myi = besti; }  // lane t owns pick t
      pv = bestv;
      pi = besti;
    }
    // softmax over the 32 picks (lane t holds pick t); pick 0 is the row max
    float m0 = __shfl(myv, 0, 32);
    float e = (myv == NEG_INF) ? 0.f : __expf(myv - m0);
    if (myv == NEG_INF) myi = 0;
    float s = e;
    for (int off = 16; off; off >>= 1) s += __shfl_xor(s, off, 32);
    float att = e / s;
    // sparse y[rr,:] = sum_t att_t * V[idx_t,:]; lane covers dims {lane,lane+32}
    float acc0 = 0.f, acc1 = 0.f;
    for (int t = 0; t < 32; ++t) {
      float w = __shfl(att, t, 32);
      int id = __shfl(myi, t, 32);
      if (w != 0.f) {  // wave-uniform branch
        const float* vrow = Vbh + (size_t)id * VD;
        acc0 += w * vrow[lane];
        acc1 += w * vrow[lane + 32];
      }
    }
    size_t ybase = ((size_t)b * Sc + q0 + rr) * (Hc * VD) + (size_t)h * VD;
    Yb[ybase + lane] = f32_to_bf16(acc0);
    Yb[ybase + lane + 32] = f32_to_bf16(acc1);
  }
}

// ---------------------------------------------------------------- kernel 3
// out = y @ Wo + bo, f32 out. grid (B*S/16, 8), 4 waves x 16-col tiles.
__global__ __launch_bounds__(128) void out_proj_kernel(
    const unsigned short* __restrict__ Yb, const unsigned short* __restrict__ WoT,
    const float* __restrict__ bo, float* __restrict__ out) {
  int row0 = blockIdx.x * 16;
  int wave = threadIdx.x >> 5, lane = threadIdx.x & 31;
  int half = lane >> 4, r = lane & 15;
  int ct = blockIdx.y * 4 + wave, col0 = ct * 16;
  v8f c = {};
  for (int kk = 0; kk < IND; kk += 32) {
    v16bf a = load_fragA_bf16(Yb, row0, IND, kk);
    v16bf bb = load_fragB_bf16(WoT, col0, IND, kk);
    c = wmma_bf16(a, bb, c);
  }
  float bias = bo[col0 + r];
#pragma unroll
  for (int i = 0; i < 8; ++i) {
    int m = half * 8 + i;
    out[(size_t)(row0 + m) * OUTD + col0 + r] = c[i] + bias;
  }
}

// ---------------------------------------------------------------- launcher
extern "C" void kernel_launch(void* const* d_in, const int* in_sizes, int n_in,
                              void* d_out, int out_size, void* d_ws, size_t ws_size,
                              hipStream_t stream) {
  const float* x  = (const float*)d_in[0];
  const float* Wk = (const float*)d_in[1];
  const float* bk = (const float*)d_in[2];
  const float* Wv = (const float*)d_in[3];
  const float* bv = (const float*)d_in[4];
  const float* Wo = (const float*)d_in[5];
  const float* bo = (const float*)d_in[6];
  float* out = (float*)d_out;

  char* ws = (char*)d_ws;
  size_t off = 0;
  auto carve = [&](size_t bytes) {
    void* p = ws + off;
    off = (off + bytes + 255) & ~(size_t)255;
    return p;
  };
  unsigned short* WkT = (unsigned short*)carve((size_t)Hc * KD * IND * 2);
  unsigned short* WvT = (unsigned short*)carve((size_t)Hc * KD * IND * 2);
  unsigned short* WoT = (unsigned short*)carve((size_t)OUTD * IND * 2);
  unsigned short* xbb = (unsigned short*)carve((size_t)Bc * Sc * IND * 2);
  unsigned short* Kb  = (unsigned short*)carve((size_t)Bc * Hc * Sc * KD * 2);
  float*          Vf  = (float*)carve((size_t)Bc * Hc * Sc * VD * 4);
  unsigned short* Yb  = (unsigned short*)carve((size_t)Bc * Sc * Hc * VD * 2);
  (void)in_sizes; (void)n_in; (void)out_size; (void)ws_size;

  // 0a) weight prep: 262144 elements
  prep_weights_kernel<<<(Hc * KD * IND) / 256, 256, 0, stream>>>(Wk, Wv, Wo, WkT,
                                                                 WvT, WoT);
  // 0b) x -> bf16 (float4-vectorized)
  convert_x_kernel<<<((size_t)Bc * Sc * IND / 4) / 256, 256, 0, stream>>>(x, xbb);
  // 1) K/V projection
  dim3 g1(Bc * Hc, Sc / 16);
  proj_kv_kernel<<<g1, 256, 0, stream>>>(xbb, WkT, WvT, bk, bv, Kb, Vf);
  // 2) attention with top-32 sparsification (128 KB score strip + 2 KB TDM tile)
  size_t lds2 = (size_t)16 * Sc * sizeof(float) + 16 * KD * sizeof(unsigned short);
  attn_kernel<<<Bc * Hc * (Sc / 16), 128, lds2, stream>>>(Kb, Vf, Yb);
  // 3) output projection
  dim3 g3((Bc * Sc) / 16, OUTD / 64);
  out_proj_kernel<<<g3, 128, 0, stream>>>(Yb, WoT, bo, out);
}
